// ODConv2d_74062416053349
// MI455X (gfx1250) — matrix operand order repaired and checked by reading, not compiled, not run on hardware
//
#include <hip/hip_runtime.h>
#include <math.h>

// ---------------- CDNA5 WMMA types ----------------
typedef __bf16  v16bf __attribute__((ext_vector_type(16)));
typedef float   v8f   __attribute__((ext_vector_type(8)));

union FragA {
    uint4 u[2];
    v16bf v;
};

// Problem constants
#define Bn    32
#define CIN   64
#define COUT  64
#define Hh    128
#define Ww    128
#define HID   16
#define Kk    4
#define ROWS  8          // output rows per workgroup
#define FRAG_PER_SAMPLE 36864   // 4 ob * 18 slices * 32 lanes * 16 elems (bf16)

// ---------------- Kernel 1a: global average pool ----------------
__global__ __launch_bounds__(256)
void pool_kernel(const float* __restrict__ x, float* __restrict__ g) {
    __shared__ float red[256];
    const int bc = blockIdx.x;                 // b*64 + c
    const float* p = x + (size_t)bc * (Hh * Ww);
    float s = 0.f;
    for (int i = threadIdx.x; i < Hh * Ww; i += 256) s += p[i];
    red[threadIdx.x] = s;
    __syncthreads();
    for (int st = 128; st > 0; st >>= 1) {
        if (threadIdx.x < st) red[threadIdx.x] += red[threadIdx.x + st];
        __syncthreads();
    }
    if (threadIdx.x == 0) g[bc] = red[0] * (1.0f / (Hh * Ww));
}

// ---------------- Kernel 1b: attention heads ----------------
__global__ __launch_bounds__(64)
void attn_kernel(const float* __restrict__ g,
                 const float* __restrict__ mlp_w, const float* __restrict__ mlp_b,
                 const float* __restrict__ wK, const float* __restrict__ bK,
                 const float* __restrict__ wO, const float* __restrict__ bO,
                 const float* __restrict__ wI, const float* __restrict__ bI,
                 const float* __restrict__ wH, const float* __restrict__ bH,
                 const float* __restrict__ wW, const float* __restrict__ bW,
                 float* __restrict__ aK, float* __restrict__ aO,
                 float* __restrict__ aI, float* __restrict__ aH,
                 float* __restrict__ aWt) {
    const int b = blockIdx.x;
    const int t = threadIdx.x;
    __shared__ float sg[CIN], sh[HID], sk[Kk], sH[3], sW3[3];

    sg[t] = g[b * CIN + t];
    __syncthreads();

    if (t < HID) {
        float a = mlp_b[t];
        for (int c = 0; c < CIN; ++c) a += sg[c] * mlp_w[t * CIN + c];
        sh[t] = a > 0.f ? a : 0.f;
    }
    __syncthreads();

    {   // aO / aI : 64 wide
        float ao = bO[t], ai = bI[t];
        for (int j = 0; j < HID; ++j) {
            ao += sh[j] * wO[t * HID + j];
            ai += sh[j] * wI[t * HID + j];
        }
        aO[b * COUT + t] = 1.f / (1.f + expf(-ao));
        aI[b * CIN  + t] = 1.f / (1.f + expf(-ai));
    }
    if (t < Kk) {
        float lk = bK[t];
        for (int j = 0; j < HID; ++j) lk += sh[j] * wK[t * HID + j];
        sk[t] = lk;
    }
    if (t < 3) {
        float lh = bH[t], lw = bW[t];
        for (int j = 0; j < HID; ++j) {
            lh += sh[j] * wH[t * HID + j];
            lw += sh[j] * wW[t * HID + j];
        }
        sH[t]  = 1.f / (1.f + expf(-lh));
        sW3[t] = 1.f / (1.f + expf(-lw));
    }
    __syncthreads();
    if (t == 0) {
        float m = fmaxf(fmaxf(sk[0], sk[1]), fmaxf(sk[2], sk[3]));
        float e0 = expf(sk[0] - m), e1 = expf(sk[1] - m);
        float e2 = expf(sk[2] - m), e3 = expf(sk[3] - m);
        float sum = e0 + e1 + e2 + e3;
        aK[b * Kk + 0] = e0 / sum; aK[b * Kk + 1] = e1 / sum;
        aK[b * Kk + 2] = e2 / sum; aK[b * Kk + 3] = e3 / sum;
        float hs = sH[0] + sH[1] + sH[2] + 1e-6f;
        float ws = sW3[0] + sW3[1] + sW3[2] + 1e-6f;
        for (int i = 0; i < 3; ++i) {
            aH[b * 3 + i]  = sH[i] / hs;
            aWt[b * 3 + i] = sW3[i] / ws;
        }
    }
}

// ---------------- Kernel 2: build dynamic weights in WMMA A-fragment layout ----------------
// flat index f = ((ob*18 + s)*32 + lane)*16 + e  per sample
// s = (kh*3+kw)*2 + chalf ; A 16x32 bf16 layout: lane = g*16+m,
//   element e -> k' = g*16 + e, c = chalf*32 + k', o = ob*16+m
__global__ __launch_bounds__(256)
void build_wdyn_kernel(const float* __restrict__ weight,
                       const float* __restrict__ aK, const float* __restrict__ aO,
                       const float* __restrict__ aI, const float* __restrict__ aH,
                       const float* __restrict__ aWt,
                       __bf16* __restrict__ wA) {
    const int b = blockIdx.x / 144;
    const int f = (blockIdx.x % 144) * 256 + threadIdx.x;   // 0..36863
    const int e  = f & 15;
    const int L  = (f >> 4) & 31;
    const int su = f >> 9;
    const int s  = su % 18;
    const int ob = su / 18;
    const int g  = L >> 4;
    const int m  = L & 15;
    const int pos   = s >> 1;
    const int chalf = s & 1;
    const int kh = pos / 3, kw = pos % 3;
    const int kp = (g << 4) + e;
    const int c  = chalf * 32 + kp;
    const int o  = ob * 16 + m;

    float wsum = 0.f;
#pragma unroll
    for (int k = 0; k < Kk; ++k)
        wsum += aK[b * Kk + k] *
                weight[(((size_t)(k * COUT + o) * CIN + c) * 9) + kh * 3 + kw];

    float val = wsum * aO[b * COUT + o] * aI[b * CIN + c] *
                aH[b * 3 + kh] * aWt[b * 3 + kw];
    wA[(size_t)b * FRAG_PER_SAMPLE + f] = (__bf16)val;
}

// ---------------- Kernel 3: implicit-GEMM conv via V_WMMA_F32_16X16X32_BF16 ----------------
__global__ __launch_bounds__(256)
void conv_wmma_kernel(const float* __restrict__ x, const __bf16* __restrict__ wA,
                      const float* __restrict__ bias, float* __restrict__ out) {
    __shared__ __align__(32) uint4 sW4[FRAG_PER_SAMPLE / 8];   // 73728 B of bf16 fragments
    __shared__ float sBias[COUT];

    const int b  = blockIdx.x / (Hh / ROWS);
    const int y0 = (blockIdx.x % (Hh / ROWS)) * ROWS;
    const int tid = threadIdx.x;

    // stage A-fragment bank for this sample into LDS (coalesced b128)
    {
        const uint4* src = reinterpret_cast<const uint4*>(wA + (size_t)b * FRAG_PER_SAMPLE);
        for (int i = tid; i < FRAG_PER_SAMPLE / 8; i += 256) sW4[i] = src[i];
        if (tid < COUT) sBias[tid] = bias[tid];
    }
    __syncthreads();

    const int wave = tid >> 5;
    const int lane = tid & 31;
    const int g    = lane >> 4;      // lane group
    const int nl   = lane & 15;      // pixel within tile / N index
    const int x0   = wave * 16;      // 8 waves cover one 128-wide row

    for (int y = y0; y < y0 + ROWS; ++y) {
        v8f acc[4];
#pragma unroll
        for (int ob = 0; ob < 4; ++ob) acc[ob] = (v8f){0.f,0.f,0.f,0.f,0.f,0.f,0.f,0.f};

        for (int kh = 0; kh < 3; ++kh) {
            const int yy = y + kh - 1;
            if (yy < 0 || yy >= Hh) continue;        // uniform across wave
            for (int kw = 0; kw < 3; ++kw) {
                const int xx    = x0 + nl + kw - 1;
                const bool colok = (xx >= 0) && (xx < Ww);
                const int xxc   = colok ? xx : 0;
                const int pos   = kh * 3 + kw;
#pragma unroll
                for (int chalf = 0; chalf < 2; ++chalf) {
                    const int s     = pos * 2 + chalf;
                    const int cbase = chalf * 32 + g * 16;   // B 32x16: element e -> K=g*16+e
                    const float* xp = x + ((size_t)(b * CIN + cbase) * (Hh * Ww)
                                           + (size_t)yy * Ww + xxc);
                    v16bf bf;
#pragma unroll
                    for (int e = 0; e < 16; ++e) {
                        float v = colok ? xp[e * (Hh * Ww)] : 0.0f;
                        bf[e] = (__bf16)v;
                    }
#pragma unroll
                    for (int ob = 0; ob < 4; ++ob) {
                        FragA fa;
                        const int fi = ((ob * 18 + s) * 32 + lane) * 2;
                        fa.u[0] = sW4[fi];
                        fa.u[1] = sW4[fi + 1];
                        acc[ob] = __builtin_amdgcn_wmma_f32_16x16x32_bf16(
                            /*neg_a=*/false, fa.v,
                            /*neg_b=*/false, bf,
                            /*c_mod=*/(short)0, acc[ob],
                            /*reuse_a=*/false, /*reuse_b=*/false);
                    }
                }
            }
        }

        // C/D layout: VGPR r, lanes 0-15 -> M=r, lanes 16-31 -> M=8+r; N=lane%16
#pragma unroll
        for (int ob = 0; ob < 4; ++ob) {
#pragma unroll
            for (int r = 0; r < 8; ++r) {
                const int o = ob * 16 + g * 8 + r;
                const float v = acc[ob][r] + sBias[o];
                float* dst = out + (((size_t)(b * COUT + o) * Hh + y) * Ww + x0 + nl);
                __builtin_nontemporal_store(v, dst);
            }
        }
    }
}

// ---------------- Launcher ----------------
extern "C" void kernel_launch(void* const* d_in, const int* in_sizes, int n_in,
                              void* d_out, int out_size, void* d_ws, size_t ws_size,
                              hipStream_t stream) {
    const float* x      = (const float*)d_in[0];
    const float* weight = (const float*)d_in[1];
    const float* bias   = (const float*)d_in[2];
    const float* mlp_w  = (const float*)d_in[3];
    const float* mlp_b  = (const float*)d_in[4];
    const float* wK = (const float*)d_in[5];  const float* bK = (const float*)d_in[6];
    const float* wO = (const float*)d_in[7];  const float* bO = (const float*)d_in[8];
    const float* wI = (const float*)d_in[9];  const float* bI = (const float*)d_in[10];
    const float* wH = (const float*)d_in[11]; const float* bH = (const float*)d_in[12];
    const float* wW = (const float*)d_in[13]; const float* bW = (const float*)d_in[14];
    float* out = (float*)d_out;

    char* ws = (char*)d_ws;
    float* g   = (float*)ws;          // 2048
    float* aK  = g   + 2048;          // 128
    float* aO  = aK  + 128;           // 2048
    float* aI  = aO  + 2048;          // 2048
    float* aH  = aI  + 2048;          // 96
    float* aWt = aH  + 96;            // 96
    __bf16* wA = (__bf16*)(ws + 32768);  // 32*36864 bf16 = 2.36 MB

    pool_kernel<<<Bn * CIN, 256, 0, stream>>>(x, g);
    attn_kernel<<<Bn, 64, 0, stream>>>(g, mlp_w, mlp_b, wK, bK, wO, bO,
                                       wI, bI, wH, bH, wW, bW,
                                       aK, aO, aI, aH, aWt);
    build_wdyn_kernel<<<Bn * 144, 256, 0, stream>>>(weight, aK, aO, aI, aH, aWt, wA);
    conv_wmma_kernel<<<Bn * (Hh / ROWS), 256, 0, stream>>>(x, wA, bias, out);
}
